// VectorQuantizer_51153060495520
// MI455X (gfx1250) — matrix operand order repaired and checked by reading, not compiled, not run on hardware
//
#include <hip/hip_runtime.h>
#include <hip/hip_bf16.h>

typedef __attribute__((ext_vector_type(8)))  _Float16     v8h;
typedef __attribute__((ext_vector_type(16))) _Float16     v16h;
typedef __attribute__((ext_vector_type(8)))  float        v8f;
typedef __attribute__((ext_vector_type(4))) unsigned int  u32x4;
typedef __attribute__((ext_vector_type(8)))  int          i32x8;
typedef __attribute__((ext_vector_type(4)))  int          i32x4;

#define N_ROWS    32768              // B*H*W
#define C_DIM     256
#define K_CODES   1024
#define OUT_ELEMS (32*256*32*32)     // 8388608

#define CHUNK     64                 // codes per LDS chunk
#define NCHUNK    (K_CODES / CHUNK)  // 16
#define ROWP      264                // padded LDS row in halfs (528 B)

#if defined(__has_builtin)
#if __has_builtin(__builtin_amdgcn_tensor_load_to_lds) && __has_builtin(__builtin_amdgcn_s_wait_tensorcnt)
#define USE_TDM 1
#endif
#endif
#ifndef USE_TDM
#define USE_TDM 0
#endif

// ---------------------------------------------------------------------------
// Kernel 1: NCHW f32 -> [N, C] row-major f16 (batched 256x1024 transpose)
// ---------------------------------------------------------------------------
__global__ __launch_bounds__(256) void vq_transpose_kernel(
    const float* __restrict__ x, _Float16* __restrict__ zf)
{
  __shared__ float tile[32][33];
  const int b  = blockIdx.z;
  const int c0 = blockIdx.x * 32;   // channel tile
  const int p0 = blockIdx.y * 32;   // hw tile
  const int tx = threadIdx.x;       // 32
  const int ty = threadIdx.y;       // 8
#pragma unroll
  for (int j = 0; j < 4; ++j)
    tile[ty + 8*j][tx] = x[((b*256 + c0 + ty + 8*j) << 10) + p0 + tx];
  __syncthreads();
#pragma unroll
  for (int j = 0; j < 4; ++j)
    zf[((b << 10) + p0 + ty + 8*j) * C_DIM + c0 + tx] =
        (_Float16)tile[tx][ty + 8*j];
}

// ---------------------------------------------------------------------------
// Kernel 2: codebook f32 -> f16 + per-code squared norms; zero loss accum
// ---------------------------------------------------------------------------
__global__ __launch_bounds__(256) void vq_cbprep_kernel(
    const float* __restrict__ cb, _Float16* __restrict__ cbf,
    float* __restrict__ eNorm, float* __restrict__ accum)
{
  const int k = blockIdx.x;
  const int t = threadIdx.x;
  if (k == 0 && t == 0) *accum = 0.0f;
  const float v = cb[k * C_DIM + t];
  cbf[k * C_DIM + t] = (_Float16)v;
  __shared__ float red[256];
  red[t] = v * v;
  __syncthreads();
#pragma unroll
  for (int s = 128; s > 0; s >>= 1) {
    if (t < s) red[t] += red[t + s];
    __syncthreads();
  }
  if (t == 0) eNorm[k] = red[0];
}

// ---------------------------------------------------------------------------
// TDM: DMA one 64-code chunk (contiguous 32 KB) into LDS with row padding.
// D# per CDNA5 ISA: 1-D tile, data_size=4B,
// tensor_dim0 = tile_dim0 = 8192 dwords, pad every 128 dwords by 4 dwords.
// ---------------------------------------------------------------------------
#if USE_TDM
__device__ __forceinline__ void tdm_load_chunk(const _Float16* gsrc,
                                               unsigned lds_byte) {
  const unsigned long long ga = (unsigned long long)(uintptr_t)gsrc;
  u32x4 g0;
  g0[0] = 1u;                                         // count=1, user mode
  g0[1] = lds_byte;                                   // D#.lds_addr
  g0[2] = (unsigned)(ga & 0xffffffffu);               // global_addr[31:0]
  g0[3] = (unsigned)((ga >> 32) & 0x01ffffffu)        // global_addr[56:32]
        | 0x80000000u;                                // type=2 (image)
  i32x8 g1;
  g1[0] = (int)((2u << 16)      // data_size = 4 bytes
              | (1u << 20)      // pad_enable
              | (6u << 22)      // pad_interval: 128 dwords
              | (3u << 25));    // pad_amount: 4 dwords (16 B per 512 B row)
  g1[1] = (int)0x20000000;      // tensor_dim0[15:0]=0x2000 in bits[31:16]
  g1[2] = 0;                    // tensor_dim0 hi / tensor_dim1 lo
  g1[3] = (int)0x20000000;      // tile_dim0 = 0x2000 dwords in bits[31:16]
  g1[4] = 0;                    // tile_dim1=0 (unused), tile_dim2=0
  g1[5] = 0x2000;               // tensor_dim0_stride
  g1[6] = 0;
  g1[7] = 0;
  const i32x4 z4 = {0, 0, 0, 0};
  const i32x8 z8 = {0, 0, 0, 0, 0, 0, 0, 0};
  __builtin_amdgcn_tensor_load_to_lds(g0, g1, z4, z4, z8, 0);
}
#else
__device__ __forceinline__ void copy_chunk_fallback(
    const _Float16* __restrict__ gsrc, _Float16* __restrict__ dst, int tid) {
#pragma unroll
  for (int k = 0; k < 8; ++k) {
    const int h = (tid + k * 256) * 8;       // half index in source chunk
    const int row = h >> 8, col = h & 255;
    *(v8h*)(dst + row * ROWP + col) = *(const v8h*)(gsrc + h);
  }
}
#endif

// ---------------------------------------------------------------------------
// Kernel 3: WMMA distance GEMM + fused argmin, double-buffered TDM staging.
// One wave = 16 rows; 8 waves/block share the LDS codebook chunk.
// score(n,k) = ||e_k||^2 - 2 * z_n . e_k   (||z||^2 constant in k -> dropped)
// ---------------------------------------------------------------------------
__global__ __launch_bounds__(256) void vq_argmin_kernel(
    const _Float16* __restrict__ zf,   // [N, 256]
    const _Float16* __restrict__ cbf,  // [K, 256]
    const float*    __restrict__ eNorm,// [K]
    int*            __restrict__ enc)  // [N]
{
  __shared__ _Float16 lds_cb[2][CHUNK * ROWP];   // 2 x 33792 B

  const int lane  = threadIdx.x & 31;
  const int wave  = threadIdx.x >> 5;
  const int mBase = blockIdx.x * 128 + wave * 16;
  const int lo16  = lane & 15;
  const int hiGrp = lane >> 4;   // 0 or 1

  // --- preload A: row (mBase + lo16), K=256 in 8 chunks of 32.
  // ISA 16-bit A 16x32 layout: lanes 0-15 hold K {0..7,16..23}, lanes 16-31
  // hold K {8..15,24..31} -> two contiguous b128 loads per chunk.
  const _Float16* zrow = zf + (size_t)(mBase + lo16) * C_DIM;
  v16h A[8];
#pragma unroll
  for (int kc = 0; kc < 8; ++kc) {
    const int base = kc * 32 + hiGrp * 8;
    v8h a0 = *(const v8h*)(zrow + base);
    v8h a1 = *(const v8h*)(zrow + base + 16);
#pragma unroll
    for (int e = 0; e < 8; ++e) { A[kc][e] = a0[e]; A[kc][e + 8] = a1[e]; }
  }

  float bestVal[8];
  int   bestIdx[8];
#pragma unroll
  for (int i = 0; i < 8; ++i) { bestVal[i] = 3.4e38f; bestIdx[i] = 0; }

  // ---- stage chunk 0 ----
#if USE_TDM
  if (threadIdx.x < 32) {
    tdm_load_chunk(cbf, (unsigned)(uintptr_t)(const void*)&lds_cb[0][0]);
    __builtin_amdgcn_s_wait_tensorcnt(0);
  }
#else
  copy_chunk_fallback(cbf, &lds_cb[0][0], threadIdx.x);
#endif
  __syncthreads();

  for (int ch = 0; ch < NCHUNK; ++ch) {
    const int p = ch & 1;
    // kick off the next chunk into the back buffer (overlaps WMMA below)
#if USE_TDM
    if ((ch + 1 < NCHUNK) && (threadIdx.x < 32))
      tdm_load_chunk(cbf + (size_t)(ch + 1) * CHUNK * C_DIM,
                     (unsigned)(uintptr_t)(const void*)&lds_cb[p ^ 1][0]);
#else
    if (ch + 1 < NCHUNK)
      copy_chunk_fallback(cbf + (size_t)(ch + 1) * CHUNK * C_DIM,
                          &lds_cb[p ^ 1][0], threadIdx.x);
#endif

    const _Float16* lbase = &lds_cb[p][0];
#pragma unroll
    for (int t4 = 0; t4 < CHUNK / 16; ++t4) {
      const int codeLocal = t4 * 16 + lo16;
      const int code      = ch * CHUNK + codeLocal;
      // ISA 16-bit B 32x16 layout: lane = column; lanes 0-15 hold K {0..7,16..23},
      // lanes 16-31 hold K {8..15,24..31} -> two contiguous b128 loads per chunk.
      const _Float16* crow = lbase + codeLocal * ROWP;
      v8f acc = {};
#pragma unroll
      for (int kc = 0; kc < 8; ++kc) {
        const int cbase = kc * 32 + hiGrp * 8;
        v8h b0 = *(const v8h*)(crow + cbase);
        v8h b1 = *(const v8h*)(crow + cbase + 16);
        v16h Bm;
#pragma unroll
        for (int e = 0; e < 8; ++e) { Bm[e] = b0[e]; Bm[e + 8] = b1[e]; }
        acc = __builtin_amdgcn_wmma_f32_16x16x32_f16(
            false, A[kc], false, Bm, (short)0, acc, false, false);
      }
      const float en = eNorm[code];
#pragma unroll
      for (int i = 0; i < 8; ++i) {
        const float s = en - 2.0f * acc[i];
        if (s < bestVal[i]) { bestVal[i] = s; bestIdx[i] = code; }
      }
    }

#if USE_TDM
    if (threadIdx.x < 32) __builtin_amdgcn_s_wait_tensorcnt(0);
#endif
    __syncthreads();
  }

  // C/D layout: lane holds col N=lane%16; vgpr i is row M = i + 8*hiGrp.
  // Reduce min across the 16 lanes sharing a row (xor<16 stays in half-wave).
#pragma unroll
  for (int i = 0; i < 8; ++i) {
    float v = bestVal[i];
    int   ix = bestIdx[i];
#pragma unroll
    for (int off = 1; off < 16; off <<= 1) {
      const float ov = __shfl_xor(v, off, 32);
      const int   oi = __shfl_xor(ix, off, 32);
      if (ov < v || (ov == v && oi < ix)) { v = ov; ix = oi; }
    }
    if (lo16 == 0) enc[mBase + hiGrp * 8 + i] = ix;
  }
}

// ---------------------------------------------------------------------------
// Kernel 4: gather codebook rows -> NCHW f32 output + squared-error reduce
// ---------------------------------------------------------------------------
__global__ __launch_bounds__(256) void vq_gather_loss_kernel(
    const float* __restrict__ x, const float* __restrict__ cb,
    const int* __restrict__ enc, float* __restrict__ out,
    float* __restrict__ accum)
{
  const int tid = blockIdx.x * 256 + threadIdx.x;  // = ((b*256+c)<<10) + hw
  const int hw  = tid & 1023;
  const int bc  = tid >> 10;
  const int c   = bc & 255;
  const int b   = bc >> 8;
  const int n   = (b << 10) + hw;
  const int k   = enc[n];
  const float q = cb[k * C_DIM + c];
  const float z = x[tid];
  out[tid] = q;                 // straight-through forward value
  float d = q - z;
  d = d * d;
  __shared__ float red[256];
  red[threadIdx.x] = d;
  __syncthreads();
#pragma unroll
  for (int s = 128; s > 0; s >>= 1) {
    if (threadIdx.x < s) red[threadIdx.x] += red[threadIdx.x + s];
    __syncthreads();
  }
  if (threadIdx.x == 0) atomicAdd(accum, red[0]);
}

// ---------------------------------------------------------------------------
// Kernel 5: finalize scalars + indices (as float) into output tail
// ---------------------------------------------------------------------------
__global__ __launch_bounds__(256) void vq_finalize_kernel(
    const float* __restrict__ accum, const int* __restrict__ enc,
    float* __restrict__ tail)
{
  const int tid = blockIdx.x * 256 + threadIdx.x;
  if (tid == 0) {
    const float mse = *accum / (float)OUT_ELEMS;
    tail[0] = 1.25f * mse;   // quantizer_loss
    tail[1] = 0.25f * mse;   // e_latent_loss
    tail[2] = mse;           // q_latent_loss
  }
  if (tid < N_ROWS) tail[3 + tid] = (float)enc[tid];
}

// ---------------------------------------------------------------------------
extern "C" void kernel_launch(void* const* d_in, const int* in_sizes, int n_in,
                              void* d_out, int out_size, void* d_ws, size_t ws_size,
                              hipStream_t stream) {
  const float* x  = (const float*)d_in[0];   // [32,256,32,32]
  const float* cb = (const float*)d_in[1];   // [1024,256]
  float* out = (float*)d_out;

  char* ws = (char*)d_ws;
  _Float16* zf    = (_Float16*)(ws);                       // 16 MB
  _Float16* cbf   = (_Float16*)(ws + 16777216);            // 512 KB
  float*    eNorm = (float*)   (ws + 16777216 + 524288);   // 4 KB
  int*      enc   = (int*)     (ws + 16777216 + 524288 + 4096);   // 128 KB
  float*    accum = (float*)   (ws + 16777216 + 524288 + 4096 + 131072);

  dim3 tgrid(256 / 32, 1024 / 32, 32);
  dim3 tblk(32, 8);
  vq_transpose_kernel<<<tgrid, tblk, 0, stream>>>(x, zf);
  vq_cbprep_kernel<<<K_CODES, 256, 0, stream>>>(cb, cbf, eNorm, accum);
  vq_argmin_kernel<<<N_ROWS / 128, 256, 0, stream>>>(zf, cbf, eNorm, enc);
  vq_gather_loss_kernel<<<OUT_ELEMS / 256, 256, 0, stream>>>(x, cb, enc, out, accum);
  vq_finalize_kernel<<<128, 256, 0, stream>>>(accum, enc, out + OUT_ELEMS);
}
